// NonLocalModule_55113020342732
// MI455X (gfx1250) — matrix operand order repaired and checked by reading, not compiled, not run on hardware
//
#include <hip/hip_runtime.h>
#include <hip/hip_bf16.h>

typedef __attribute__((ext_vector_type(16))) __bf16    v16bf;
typedef __attribute__((ext_vector_type(8)))  float     v8f;
typedef __attribute__((ext_vector_type(8)))  unsigned  v8u;

constexpr int kB = 8, kC = 1024, kT = 8, kH = 28, kW = 28;
constexpr int kE = 512, kN = kT * kH * kW;               // 6272
constexpr int kHs = 14, kWs = 14, kM = kT * kHs * kWs;   // 1568
constexpr float kEps = 1e-5f;

// ---------- bf16 helpers (raw ushort storage) ----------
__device__ __forceinline__ unsigned short f2b(float f) {
  unsigned u = __float_as_uint(f);
  u += 0x7fffu + ((u >> 16) & 1u);          // RNE
  return (unsigned short)(u >> 16);
}
__device__ __forceinline__ float b2f(unsigned short h) {
  return __uint_as_float(((unsigned)h) << 16);
}

// ---------- WMMA fragment loaders (CDNA5 16-bit layouts, wave32) ----------
// A 16x32 from row-major LDS: per lane two contiguous 16B runs -> 2x ds_load_b128
__device__ __forceinline__ v16bf load_a(const unsigned short* s, int ld, int row0, int lane) {
  int m = lane & 15, half = lane >> 4;
  const unsigned short* p = s + (size_t)(row0 + m) * ld + half * 16;
  uint4 lo = *(const uint4*)(p);        // k = half*16 + 0..7  -> regs 0..3
  uint4 hi = *(const uint4*)(p + 8);    // k = half*16 + 8..15 -> regs 4..7
  v8u u = {lo.x, lo.y, lo.z, lo.w, hi.x, hi.y, hi.z, hi.w};
  return __builtin_bit_cast(v16bf, u);
}
// B 32x16 from [col][k] row-major LDS (K-matrix natural layout): 2x ds_load_b128
__device__ __forceinline__ v16bf load_bt(const unsigned short* s, int ld, int col0, int k0, int lane) {
  int n = lane & 15, half = lane >> 4;
  const unsigned short* p = s + (size_t)(col0 + n) * ld + k0 + half * 16;
  uint4 lo = *(const uint4*)(p);        // k pairs v=0..3
  uint4 hi = *(const uint4*)(p + 8);    // k pairs v=4..7
  v8u u = {lo.x, lo.y, lo.z, lo.w, hi.x, hi.y, hi.z, hi.w};
  return __builtin_bit_cast(v16bf, u);
}
// B 32x16 from packed k-pair LDS: [k/2][col] uint (lo=k even, hi=k odd): ds_load_2addr_b32
__device__ __forceinline__ v16bf load_bp(const unsigned* s, int ld, int col0, int lane) {
  int n = lane & 15, half = lane >> 4;
  const unsigned* p = s + (size_t)(half * 8) * ld + col0 + n;
  v8u u;
#pragma unroll
  for (int v = 0; v < 8; ++v) u[v] = p[(size_t)v * ld];
  return __builtin_bit_cast(v16bf, u);
}
__device__ __forceinline__ v8f wmma_bf16(v16bf a, v16bf b, v8f c) {
  return __builtin_amdgcn_wmma_f32_16x16x32_bf16(false, a, false, b, (short)0, c, false, false);
}
// pack two bf16 rows (even k in lo16, odd k in hi16), 4 columns at a time
__device__ __forceinline__ uint4 pack_pair(uint2 r0, uint2 r1) {
  return make_uint4((r0.x & 0xffffu) | (r1.x << 16),
                    (r0.x >> 16)     | (r1.x & 0xffff0000u),
                    (r0.y & 0xffffu) | (r1.y << 16),
                    (r0.y >> 16)     | (r1.y & 0xffff0000u));
}

// ---------- prep: weights to bf16, transposed for GEMM-B operand ----------
__global__ void nl_prep_weights(const float* __restrict__ wth, const float* __restrict__ wph,
                                const float* __restrict__ wg, const float* __restrict__ wo,
                                unsigned short* __restrict__ wtb, unsigned short* __restrict__ wpb,
                                unsigned short* __restrict__ wgb, unsigned short* __restrict__ wob) {
  int i = blockIdx.x * 256 + threadIdx.x;
  if (i >= kE * kC) return;
  int e = i / kC, c = i % kC;
  wtb[(size_t)c * kE + e] = f2b(wth[(size_t)e * kC + c]);  // [C][E]
  wpb[(size_t)c * kE + e] = f2b(wph[(size_t)e * kC + c]);
  wgb[(size_t)c * kE + e] = f2b(wg[(size_t)e * kC + c]);
  wob[(size_t)e * kC + c] = f2b(wo[(size_t)c * kE + e]);   // [E][C]
}

// ---------- prep: x [B][C][N] f32 -> x_bf [B][N][C] bf16 (LDS-tiled transpose) ----------
__global__ __launch_bounds__(256) void nl_x_to_bf16(const float* __restrict__ x,
                                                    unsigned short* __restrict__ xb) {
  __shared__ float tile[32][33];
  int b = blockIdx.z, n0 = blockIdx.x * 32, c0 = blockIdx.y * 32;
  int tj = threadIdx.x & 31, ti = threadIdx.x >> 5;
  const float* xs = x + (size_t)b * kC * kN;
#pragma unroll
  for (int p = 0; p < 4; ++p) {
    int i = ti + p * 8;
    tile[i][tj] = xs[(size_t)(c0 + i) * kN + n0 + tj];
  }
  __syncthreads();
  unsigned short* o = xb + (size_t)b * kN * kC;
#pragma unroll
  for (int p = 0; p < 4; ++p) {
    int i = ti + p * 8;
    o[(size_t)(n0 + i) * kC + c0 + tj] = f2b(tile[tj][i]);
  }
}

// ---------- prep: maxpool (1,2,2) on bf16, N-major ----------
__global__ void nl_maxpool(const unsigned short* __restrict__ xb,
                           unsigned short* __restrict__ xsb) {
  size_t i = (size_t)blockIdx.x * 256 + threadIdx.x;
  if (i >= (size_t)kB * kM * kC) return;
  int c = (int)(i % kC);
  size_t rest = i / kC;
  int m = (int)(rest % kM);
  int b = (int)(rest / kM);
  int t = m / (kHs * kWs);
  int hs = (m / kWs) % kHs;
  int ws = m % kWs;
  int n = t * (kH * kW) + (2 * hs) * kW + 2 * ws;
  const unsigned short* base = xb + (size_t)b * kN * kC + c;
  float v0 = b2f(base[(size_t)n * kC]);
  float v1 = b2f(base[(size_t)(n + 1) * kC]);
  float v2 = b2f(base[(size_t)(n + kW) * kC]);
  float v3 = b2f(base[(size_t)(n + kW + 1) * kC]);
  xsb[((size_t)b * kM + m) * kC + c] = f2b(fmaxf(fmaxf(v0, v1), fmaxf(v2, v3)));
}

// ---------- generic GEMM: Out[b] (RxS bf16) = Act[b](RxCin bf16) @ Wt(CinxS bf16) + bias ----------
constexpr int TLDA = 40;  // 128x32 A tile: 80B rows (16B-aligned, bank stride 20)
constexpr int TLDZ = 72;  // bf16 epilogue staging: 144B rows
__global__ __launch_bounds__(256) void nl_gemm_bf16(const unsigned short* __restrict__ Act,
                                                    const unsigned short* __restrict__ Wt,
                                                    const float* __restrict__ bias,
                                                    unsigned short* __restrict__ Out,
                                                    int R, int Cin, int S) {
  __shared__ __align__(16) unsigned short As[128 * TLDA];
  __shared__ __align__(16) unsigned Bp[16 * 64];        // packed k-pairs x cols
  __shared__ __align__(16) unsigned short Zt[128 * TLDZ];
  int b = blockIdx.z;
  int row0 = blockIdx.x * 128, col0 = blockIdx.y * 64;
  const unsigned short* A = Act + (size_t)b * R * Cin;
  unsigned short* O = Out + (size_t)b * R * S;
  int tid = threadIdx.x, lane = tid & 31, wave = tid >> 5;
  int wr = wave >> 1, wc = wave & 1;  // 4x2 wave grid -> 32x32 per wave
  v8f acc[2][2] = {};
  for (int k0 = 0; k0 < Cin; k0 += 32) {
#pragma unroll
    for (int p = 0; p < 2; ++p) {  // A: 128x32 = 512 vec8 (clamped rows, no branch)
      int v = p * 256 + tid;
      int r = v >> 2, kk = (v & 3) * 8;
      int rr = row0 + r; rr = rr < R ? rr : R - 1;
      *(uint4*)(&As[r * TLDA + kk]) = *(const uint4*)(A + (size_t)rr * Cin + k0 + kk);
    }
    {  // B: 32x64 -> packed 16x64 dwords
      int pr = tid >> 4;          // pair-row 0..15
      int cc = (tid & 15) * 4;    // 4 cols
      uint2 r0 = *(const uint2*)(Wt + (size_t)(k0 + 2 * pr) * S + col0 + cc);
      uint2 r1 = *(const uint2*)(Wt + (size_t)(k0 + 2 * pr + 1) * S + col0 + cc);
      *(uint4*)(&Bp[pr * 64 + cc]) = pack_pair(r0, r1);
    }
    __syncthreads();
    v16bf a0 = load_a(As, TLDA, wr * 32, lane);
    v16bf a1 = load_a(As, TLDA, wr * 32 + 16, lane);
    v16bf b0 = load_bp(Bp, 64, wc * 32, lane);
    v16bf b1 = load_bp(Bp, 64, wc * 32 + 16, lane);
    acc[0][0] = wmma_bf16(a0, b0, acc[0][0]);
    acc[0][1] = wmma_bf16(a0, b1, acc[0][1]);
    acc[1][0] = wmma_bf16(a1, b0, acc[1][0]);
    acc[1][1] = wmma_bf16(a1, b1, acc[1][1]);
    __syncthreads();
  }
  {  // stage bf16 tile in LDS, then coalesced b128 stores
    int half = lane >> 4, n = lane & 15;
#pragma unroll
    for (int i = 0; i < 2; ++i)
#pragma unroll
      for (int j = 0; j < 2; ++j) {
        int cl = wc * 32 + j * 16 + n;
        float bb = bias[col0 + cl];
#pragma unroll
        for (int v = 0; v < 8; ++v) {
          int r = wr * 32 + i * 16 + v + half * 8;
          Zt[r * TLDZ + cl] = f2b(acc[i][j][v] + bb);
        }
      }
  }
  __syncthreads();
#pragma unroll
  for (int t = 0; t < 4; ++t) {
    int task = t * 256 + tid;
    int r = task >> 3, cc = (task & 7) * 8;
    int gr = row0 + r;
    if (gr < R)
      *(uint4*)(O + (size_t)gr * S + col0 + cc) = *(const uint4*)(&Zt[r * TLDZ + cc]);
  }
}

// ---------- fused flash attention: O = softmax(Q K^T) V, bf16 in/out, f32 accum ----------
constexpr int LDQ = 520;   // bf16 row pad: 1040B rows (16B aligned, bank stride 4)
constexpr int LDP = 40;
constexpr int ATT_SMEM = 32 * LDQ * 2        /*Qs*/
                       + 2 * 32 * LDQ * 2    /*Ks double buffer*/
                       + 16 * 512 * 4        /*Vp packed*/
                       + 8 * 32 * 32 * 4     /*Sp partials (reused as O staging)*/
                       + 32 * 32 * 4         /*Sf*/
                       + 32 * LDP * 2        /*Ps*/
                       + 3 * 32 * 4;         /*stats*/

// CDNA5 async copy: 16B/lane global -> LDS, tracked by ASYNCcnt
__device__ __forceinline__ void async_k_chunk(const unsigned short* gbase, unsigned short* dst,
                                              unsigned char* smem, int tid) {
#pragma unroll
  for (int p = 0; p < 8; ++p) {
    int v = p * 256 + tid;
    int r = v >> 6, e = (v & 63) * 8;
    unsigned lds_off = (unsigned)((char*)(dst + r * LDQ + e) - (char*)smem);
    unsigned goff = (unsigned)((r * kE + e) * 2);
    asm volatile("global_load_async_to_lds_b128 %0, %1, %2"
                 :: "v"(lds_off), "v"(goff), "s"(gbase) : "memory");
  }
}

__global__ __launch_bounds__(256) void nl_attn(const unsigned short* __restrict__ Q,
                                               const unsigned short* __restrict__ Km,
                                               const unsigned short* __restrict__ Vm,
                                               unsigned short* __restrict__ Og) {
  extern __shared__ __align__(16) unsigned char smem[];
  unsigned short* Qs = (unsigned short*)smem;
  unsigned short* Ks0 = Qs + 32 * LDQ;                 // two 32xLDQ buffers
  unsigned* Vp = (unsigned*)(Ks0 + 2 * 32 * LDQ);      // 16 x 512 packed key-pairs
  float* Sp = (float*)(Vp + 16 * 512);
  float* Sf = Sp + 8 * 32 * 32;
  unsigned short* Ps = (unsigned short*)(Sf + 32 * 32);
  float* mrow = (float*)(Ps + 32 * LDP);
  float* lrow = mrow + 32;
  float* arow = lrow + 32;

  int b = blockIdx.y, q0 = blockIdx.x * 32;
  int tid = threadIdx.x, lane = tid & 31, wave = tid >> 5;
  int ec = wave * 64;  // E-slice owned by this wave
  const unsigned short* Qb = Q + ((size_t)b * kN + q0) * kE;
  const unsigned short* Kb = Km + (size_t)b * kM * kE;
  const unsigned short* Vb = Vm + (size_t)b * kM * kE;

  async_k_chunk(Kb, Ks0, smem, tid);  // prologue: K chunk 0 in flight
#pragma unroll
  for (int p = 0; p < 8; ++p) {       // Q tile 32x512
    int v = p * 256 + tid;
    int r = v >> 6, e = (v & 63) * 8;
    *(uint4*)(&Qs[r * LDQ + e]) = *(const uint4*)(Qb + (size_t)r * kE + e);
  }
  if (tid < 32) { mrow[tid] = -1e30f; lrow[tid] = 0.f; }

  v8f acc[2][4] = {};
  constexpr int nchunk = kM / 32;  // 49
  for (int kc = 0; kc < nchunk; ++kc) {
    int k0 = kc * 32;
    int cur = kc & 1;
    unsigned short* Kcur = Ks0 + cur * (32 * LDQ);
    __syncthreads();  // previous-iter readers of Ks[1-cur]/Vp/Ps done (also fences init)
    bool more = (kc + 1) < nchunk;
    if (more)
      async_k_chunk(Kb + (size_t)(k0 + 32) * kE, Ks0 + (1 - cur) * (32 * LDQ), smem, tid);
#pragma unroll
    for (int p = 0; p < 8; ++p) {  // V chunk: load + pack key-pairs
      int idx = p * 256 + tid;     // 16 pair-rows x 128 col-groups
      int pr = idx >> 7;
      int cc = (idx & 127) * 4;
      uint2 r0 = *(const uint2*)(Vb + (size_t)(k0 + 2 * pr) * kE + cc);
      uint2 r1 = *(const uint2*)(Vb + (size_t)(k0 + 2 * pr + 1) * kE + cc);
      *(uint4*)(&Vp[pr * 512 + cc]) = pack_pair(r0, r1);
    }
    if (more) {
      __builtin_prefetch((const char*)(Vb + (size_t)(k0 + 32) * kE) + tid * 128, 0, 1);
      asm volatile("s_wait_asynccnt 0x8" ::: "memory");  // current K done, next stays in flight
    } else {
      asm volatile("s_wait_asynccnt 0x0" ::: "memory");
    }
    __syncthreads();  // publish Kcur (all waves waited own async) + Vp
    // partial S (32x32) over this wave's E-slice [ec, ec+64)
    v8f s[2][2] = {};
#pragma unroll
    for (int kk = 0; kk < 2; ++kk) {
      int ko = ec + kk * 32;
      v16bf a0 = load_a(Qs + ko, LDQ, 0, lane);
      v16bf a1 = load_a(Qs + ko, LDQ, 16, lane);
      v16bf b0 = load_bt(Kcur, LDQ, 0, ko, lane);
      v16bf b1 = load_bt(Kcur, LDQ, 16, ko, lane);
      s[0][0] = wmma_bf16(a0, b0, s[0][0]);
      s[0][1] = wmma_bf16(a0, b1, s[0][1]);
      s[1][0] = wmma_bf16(a1, b0, s[1][0]);
      s[1][1] = wmma_bf16(a1, b1, s[1][1]);
    }
    {
      int half = lane >> 4, n = lane & 15;
      float* Wp = Sp + wave * (32 * 32);
#pragma unroll
      for (int i = 0; i < 2; ++i)
#pragma unroll
        for (int j = 0; j < 2; ++j)
#pragma unroll
          for (int v = 0; v < 8; ++v)
            Wp[(i * 16 + v + half * 8) * 32 + j * 16 + n] = s[i][j][v];
    }
    __syncthreads();
#pragma unroll
    for (int p = 0; p < 4; ++p) {  // reduce 8 partials
      int idx = p * 256 + tid;
      float sum = 0.f;
#pragma unroll
      for (int w = 0; w < 8; ++w) sum += Sp[w * (32 * 32) + idx];
      Sf[idx] = sum;
    }
    __syncthreads();
    if (tid < 32) {  // online softmax, one row per thread
      int r = tid;
      float mx = -1e30f;
      for (int k = 0; k < 32; ++k) mx = fmaxf(mx, Sf[r * 32 + k]);
      float nm = fmaxf(mrow[r], mx);
      float al = __expf(mrow[r] - nm);
      float ssum = 0.f;
      for (int k = 0; k < 32; ++k) {
        float pv = __expf(Sf[r * 32 + k] - nm);
        ssum += pv;
        Ps[r * LDP + k] = f2b(pv);
      }
      lrow[r] = lrow[r] * al + ssum;
      mrow[r] = nm;
      arow[r] = al;
    }
    __syncthreads();
    {  // rescale running O, then O += P @ V-slice
      int half = lane >> 4;
#pragma unroll
      for (int i = 0; i < 2; ++i)
#pragma unroll
        for (int v = 0; v < 8; ++v) {
          float al = arow[i * 16 + v + half * 8];
#pragma unroll
          for (int j = 0; j < 4; ++j) acc[i][j][v] *= al;
        }
      v16bf pa0 = load_a(Ps, LDP, 0, lane);
      v16bf pa1 = load_a(Ps, LDP, 16, lane);
#pragma unroll
      for (int j = 0; j < 4; ++j) {
        v16bf bv = load_bp(Vp, 512, ec + j * 16, lane);
        acc[0][j] = wmma_bf16(pa0, bv, acc[0][j]);
        acc[1][j] = wmma_bf16(pa1, bv, acc[1][j]);
      }
    }
  }
  __syncthreads();
  unsigned short* Ot = (unsigned short*)Sp;  // reuse 32KB partial-S area for O staging
  {
    int half = lane >> 4, n = lane & 15;
#pragma unroll
    for (int i = 0; i < 2; ++i)
#pragma unroll
      for (int v = 0; v < 8; ++v) {
        int r = i * 16 + v + half * 8;
        float inv = 1.0f / lrow[r];
#pragma unroll
        for (int j = 0; j < 4; ++j)
          Ot[r * 512 + ec + j * 16 + n] = f2b(acc[i][j][v] * inv);
      }
  }
  __syncthreads();
  unsigned short* Ob = Og + ((size_t)b * kN + q0) * kE;
#pragma unroll
  for (int t = 0; t < 8; ++t) {
    int task = t * 256 + tid;
    int r = task >> 6, e = (task & 63) * 8;
    *(uint4*)(Ob + (size_t)r * kE + e) = *(const uint4*)(&Ot[r * 512 + e]);
  }
}

// ---------- output GEMM + bias + BN + residual, transposed coalesced store ----------
__global__ __launch_bounds__(256) void nl_out_gemm(const unsigned short* __restrict__ Og,
                                                   const unsigned short* __restrict__ Wo,
                                                   const float* __restrict__ b_out,
                                                   const float* __restrict__ gamma,
                                                   const float* __restrict__ beta,
                                                   const float* __restrict__ mean,
                                                   const float* __restrict__ var,
                                                   const float* __restrict__ x,
                                                   float* __restrict__ out) {
  __shared__ __align__(16) unsigned short As[128 * TLDA];
  __shared__ __align__(16) unsigned Bp[16 * 64];
  __shared__ float Zs[128 * 65];
  __shared__ float sc_s[64], sh_s[64];
  int b = blockIdx.z;
  int row0 = blockIdx.x * 128, col0 = blockIdx.y * 64;
  const unsigned short* A = Og + (size_t)b * kN * kE;
  int tid = threadIdx.x, lane = tid & 31, wave = tid >> 5;
  int wr = wave >> 1, wc = wave & 1;
  v8f acc[2][2] = {};
  for (int k0 = 0; k0 < kE; k0 += 32) {
#pragma unroll
    for (int p = 0; p < 2; ++p) {
      int v = p * 256 + tid;
      int r = v >> 2, kk = (v & 3) * 8;
      *(uint4*)(&As[r * TLDA + kk]) = *(const uint4*)(A + (size_t)(row0 + r) * kE + k0 + kk);
    }
    {
      int pr = tid >> 4;
      int cc = (tid & 15) * 4;
      uint2 r0 = *(const uint2*)(Wo + (size_t)(k0 + 2 * pr) * kC + col0 + cc);
      uint2 r1 = *(const uint2*)(Wo + (size_t)(k0 + 2 * pr + 1) * kC + col0 + cc);
      *(uint4*)(&Bp[pr * 64 + cc]) = pack_pair(r0, r1);
    }
    __syncthreads();
    v16bf a0 = load_a(As, TLDA, wr * 32, lane);
    v16bf a1 = load_a(As, TLDA, wr * 32 + 16, lane);
    v16bf b0 = load_bp(Bp, 64, wc * 32, lane);
    v16bf b1 = load_bp(Bp, 64, wc * 32 + 16, lane);
    acc[0][0] = wmma_bf16(a0, b0, acc[0][0]);
    acc[0][1] = wmma_bf16(a0, b1, acc[0][1]);
    acc[1][0] = wmma_bf16(a1, b0, acc[1][0]);
    acc[1][1] = wmma_bf16(a1, b1, acc[1][1]);
    __syncthreads();
  }
  if (tid < 64) {  // per-column BN affine: z*sc + sh
    int c = col0 + tid;
    float sc = gamma[c] * rsqrtf(var[c] + kEps);
    sc_s[tid] = sc;
    sh_s[tid] = (b_out[c] - mean[c]) * sc + beta[c];
  }
  {  // stage f32 tile in LDS so the [B][C][N] store is coalesced along n
    int half = lane >> 4, n = lane & 15;
#pragma unroll
    for (int i = 0; i < 2; ++i)
#pragma unroll
      for (int j = 0; j < 2; ++j)
#pragma unroll
        for (int v = 0; v < 8; ++v)
          Zs[(wr * 32 + i * 16 + v + half * 8) * 65 + wc * 32 + j * 16 + n] = acc[i][j][v];
  }
  __syncthreads();
  const float* xb = x + (size_t)b * kC * kN;
  float* ob = out + (size_t)b * kC * kN;
#pragma unroll
  for (int p = 0; p < 32; ++p) {
    int idx = p * 256 + tid;
    int cl = idx >> 7, rl = idx & 127;
    int c = col0 + cl, r = row0 + rl;
    float z = Zs[rl * 65 + cl] * sc_s[cl] + sh_s[cl];
    ob[(size_t)c * kN + r] = z + xb[(size_t)c * kN + r];
  }
}

extern "C" void kernel_launch(void* const* d_in, const int* in_sizes, int n_in,
                              void* d_out, int out_size, void* d_ws, size_t ws_size,
                              hipStream_t stream) {
  (void)in_sizes; (void)n_in; (void)out_size; (void)ws_size;
  const float* x       = (const float*)d_in[0];
  const float* w_theta = (const float*)d_in[1];
  const float* b_theta = (const float*)d_in[2];
  const float* w_phi   = (const float*)d_in[3];
  const float* b_phi   = (const float*)d_in[4];
  const float* w_g     = (const float*)d_in[5];
  const float* b_g     = (const float*)d_in[6];
  const float* w_out   = (const float*)d_in[7];
  const float* b_out   = (const float*)d_in[8];
  const float* bn_g    = (const float*)d_in[9];
  const float* bn_b    = (const float*)d_in[10];
  const float* bn_m    = (const float*)d_in[11];
  const float* bn_v    = (const float*)d_in[12];
  float* out = (float*)d_out;

  char* ws = (char*)d_ws;
  size_t off = 0;
  auto carve = [&](size_t bytes) { char* p = ws + off; off += (bytes + 255) & ~(size_t)255; return p; };
  unsigned short* wtb = (unsigned short*)carve((size_t)kC * kE * 2);
  unsigned short* wpb = (unsigned short*)carve((size_t)kC * kE * 2);
  unsigned short* wgb = (unsigned short*)carve((size_t)kC * kE * 2);
  unsigned short* wob = (unsigned short*)carve((size_t)kE * kC * 2);
  unsigned short* xb  = (unsigned short*)carve((size_t)kB * kN * kC * 2);
  unsigned short* xsb = (unsigned short*)carve((size_t)kB * kM * kC * 2);
  unsigned short* Qb  = (unsigned short*)carve((size_t)kB * kN * kE * 2);
  unsigned short* Kb  = (unsigned short*)carve((size_t)kB * kM * kE * 2);
  unsigned short* Vb  = (unsigned short*)carve((size_t)kB * kM * kE * 2);
  unsigned short* Ob  = (unsigned short*)carve((size_t)kB * kN * kE * 2);

  nl_prep_weights<<<(kE * kC + 255) / 256, 256, 0, stream>>>(w_theta, w_phi, w_g, w_out,
                                                             wtb, wpb, wgb, wob);
  nl_x_to_bf16<<<dim3(kN / 32, kC / 32, kB), 256, 0, stream>>>(x, xb);
  nl_maxpool<<<(int)(((size_t)kB * kM * kC) / 256), 256, 0, stream>>>(xb, xsb);

  nl_gemm_bf16<<<dim3((kN + 127) / 128, kE / 64, kB), 256, 0, stream>>>(xb, wtb, b_theta, Qb, kN, kC, kE);
  nl_gemm_bf16<<<dim3((kM + 127) / 128, kE / 64, kB), 256, 0, stream>>>(xsb, wpb, b_phi, Kb, kM, kC, kE);
  nl_gemm_bf16<<<dim3((kM + 127) / 128, kE / 64, kB), 256, 0, stream>>>(xsb, wgb, b_g, Vb, kM, kC, kE);

  nl_attn<<<dim3(kN / 32, kB), 256, ATT_SMEM, stream>>>(Qb, Kb, Vb, Ob);

  nl_out_gemm<<<dim3(kN / 128, kC / 64, kB), 256, 0, stream>>>(Ob, wob, b_out, bn_g, bn_b,
                                                               bn_m, bn_v, x, out);
}
